// MultiHeadAttention_58866821759418
// MI455X (gfx1250) — matrix operand order repaired and checked by reading, not compiled, not run on hardware
//
#include <hip/hip_runtime.h>
#include <hip/hip_bf16.h>
#include <stdint.h>

#define D_MODEL 1024
#define NHEADS  16
#define DEPTH   64
#define SEQ     2048
#define BATCH   2
#define MROWS   (BATCH*SEQ)      // 4096
#define NEGINF  (-1e9f)

typedef __bf16 bf16_t;
typedef __attribute__((ext_vector_type(8)))  __bf16 v8bf;
typedef __attribute__((ext_vector_type(16))) __bf16 v16bf;
typedef __attribute__((ext_vector_type(8)))  float  v8f;

// ---- WMMA helpers -----------------------------------------------------------

static __device__ __forceinline__ v16bf cat8(v8bf lo, v8bf hi) {
  return __builtin_shufflevector(lo, hi, 0,1,2,3,4,5,6,7,8,9,10,11,12,13,14,15);
}

static __device__ __forceinline__ v8f wmma_bf16(v16bf a, v16bf b, v8f c) {
  // D = A(16x32 bf16) x B(32x16 bf16) + C(16x16 f32)
  return __builtin_amdgcn_wmma_f32_16x16x32_bf16(
      /*neg_a=*/false, a, /*neg_b=*/false, b,
      /*c_mod=*/(short)0, c, /*reuse_a=*/false, /*reuse_b=*/false);
}

// A fragment: lane holds row = lane%16; elements j: k = (lane/16)*16 + j
// -> two contiguous 16B chunks at (k + halfw*16) and (k + halfw*16 + 8)
static __device__ __forceinline__ v16bf load_afrag(const bf16_t* rowp_kb, int halfw) {
  const bf16_t* p = rowp_kb + halfw*16;
  v8bf lo = *(const v8bf*)p;
  v8bf hi = *(const v8bf*)(p + 8);
  return cat8(lo, hi);
}

// B fragment from a [N,K]-major buffer (row n = column n of B): lane holds
// col = lane%16; elements j: k = (lane/16)*16 + j -> 16 contiguous bf16.
static __device__ __forceinline__ v16bf load_bfrag(const bf16_t* colrow_kb_half) {
  v8bf lo = *(const v8bf*)colrow_kb_half;
  v8bf hi = *(const v8bf*)(colrow_kb_half + 8);
  return cat8(lo, hi);
}

// ---- precision conversion / weight transpose --------------------------------

__global__ void cvt_f32_to_bf16(const float* __restrict__ in, bf16_t* __restrict__ out, int n) {
  int i = blockIdx.x * blockDim.x + threadIdx.x;
  if (i < n) out[i] = (bf16_t)in[i];
}

__global__ void transpose_cvt_1k(const float* __restrict__ W, bf16_t* __restrict__ WT) {
  // W: [K=1024, N=1024] fp32 row-major -> WT: [N,K] bf16
  int idx = blockIdx.x * blockDim.x + threadIdx.x;
  int nn = idx >> 10, kk = idx & 1023;
  WT[idx] = (bf16_t)W[(size_t)kk * D_MODEL + nn];
}

// ---- generic projection GEMM: Y = X @ W + b ---------------------------------
// X [4096,1024] bf16 row-major, WT [1024(N),1024(K)] bf16, bias fp32[1024]
// mode 0: out bf16 head layout  Qh/Kh [B,H,S,d]
// mode 1: out bf16 transposed   VhT   [B,H,d,S]
// mode 2: out fp32 row-major [4096,1024]
__global__ void __launch_bounds__(256)
gemm_qkvo(const bf16_t* __restrict__ X, const bf16_t* __restrict__ WT,
          const float* __restrict__ bias, void* __restrict__ out, int mode)
{
  const int wave  = threadIdx.x >> 5;
  const int lane  = threadIdx.x & 31;
  const int halfw = lane >> 4;
  const int l16   = lane & 15;
  const int mBase = blockIdx.y * 32 + (wave >> 2) * 16;
  const int nBase = blockIdx.x * 64 + (wave & 3) * 16;

  const bf16_t* Arow = X  + (size_t)(mBase + l16) * D_MODEL;
  const bf16_t* Brow = WT + (size_t)(nBase + l16) * D_MODEL;

  v8f acc = {};
  for (int kb = 0; kb < D_MODEL; kb += 32) {
    if (kb + 64 <= D_MODEL) {
      __builtin_prefetch(Arow + kb + 64, 0, 1);   // -> global_prefetch_b8
      __builtin_prefetch(Brow + kb + 64, 0, 1);
    }
    v16bf a = load_afrag(Arow + kb, halfw);
    v16bf b = load_bfrag(Brow + kb + halfw * 16);
    acc = wmma_bf16(a, b, acc);
  }

  const int n = nBase + l16;
  const float bn = bias[n];
#pragma unroll
  for (int i = 0; i < 8; ++i) {
    const int m = mBase + halfw * 8 + i;
    const float vA = acc[i] + bn;
    if (mode == 2) {
      ((float*)out)[(size_t)m * D_MODEL + n] = vA;
    } else {
      const int bI = m >> 11;            // /SEQ
      const int s  = m & (SEQ - 1);
      const int h  = n >> 6;             // /DEPTH
      const int dd = n & 63;
      const bf16_t bv = (bf16_t)vA;
      if (mode == 0)
        ((bf16_t*)out)[(((size_t)(bI * NHEADS + h)) * SEQ + s) * DEPTH + dd] = bv;
      else
        ((bf16_t*)out)[(((size_t)(bI * NHEADS + h)) * DEPTH + dd) * SEQ + s] = bv;
    }
  }
}

// ---- fused attention: scores + rpe + masks + softmax + P@V ------------------
// One block = one (b, h, 16-query stripe). 16x2048 fp32 score stripe lives in
// LDS (132KB of the 320KB/WGP). 8 waves, 256 threads.
__global__ void __launch_bounds__(256)
attn_fused(const bf16_t* __restrict__ Qh, const bf16_t* __restrict__ Kh,
           const bf16_t* __restrict__ VhT, const uint8_t* __restrict__ mask,
           float* __restrict__ attnOut, bf16_t* __restrict__ Ctx)
{
  __shared__ float Pb[16][SEQ + 8];      // 131,584 B score/prob stripe
  __shared__ float partC[4][16][16];     // 4,096 B split-K partials

  const int b = blockIdx.z, h = blockIdx.y, qt = blockIdx.x;
  const int qbase = qt * 16;
  const int wave  = threadIdx.x >> 5;
  const int lane  = threadIdx.x & 31;
  const int halfw = lane >> 4;
  const int l16   = lane & 15;
  const size_t head = (size_t)b * NHEADS + h;
  const float scale = 0.125f;            // 1/sqrt(64)

  // Q fragments (16 x 64, two K=32 slabs), loaded once
  v16bf qa0, qa1;
  {
    const bf16_t* Qrow = Qh + (head * SEQ + qbase + l16) * DEPTH;
    qa0 = load_afrag(Qrow, halfw);
    qa1 = load_afrag(Qrow + 32, halfw);
  }

  // ---- phase 1: scores = Q K^T * scale + rpe, masked, into LDS
  for (int t = 0; t < 16; ++t) {
    const int jt   = wave + 8 * t;       // key tile 0..127
    const int keyT = jt * 16;
    const bf16_t* Krow = Kh + (head * SEQ + keyT + l16) * DEPTH;
    v8f c = {};
    c = wmma_bf16(qa0, load_bfrag(Krow + halfw * 16), c);
    c = wmma_bf16(qa1, load_bfrag(Krow + 32 + halfw * 16), c);
    const int key = keyT + l16;
#pragma unroll
    for (int i = 0; i < 8; ++i) {
      const int r    = halfw * 8 + i;
      const int qrow = qbase + r;
      float val = c[i] * scale + (float)(key - qrow);
      const bool mk = mask[((size_t)b * SEQ + qrow) * SEQ + key] != 0;
      if (key > qrow || mk) val = NEGINF;
      Pb[r][key] = val;
    }
  }
  __syncthreads();

  // ---- phase 2: rowwise softmax (2 rows per wave), write probs to attn out
#pragma unroll
  for (int rr = 0; rr < 2; ++rr) {
    const int r = wave * 2 + rr;
    float* Pr = &Pb[r][0];
    float mx = -3.0e38f;
    for (int k0 = lane; k0 < SEQ; k0 += 32) mx = fmaxf(mx, Pr[k0]);
    for (int off = 16; off >= 1; off >>= 1) mx = fmaxf(mx, __shfl_xor(mx, off, 32));
    float sum = 0.f;
    for (int k0 = lane; k0 < SEQ; k0 += 32) {
      const float e = __expf(Pr[k0] - mx);
      Pr[k0] = e;
      sum += e;
    }
    for (int off = 16; off >= 1; off >>= 1) sum += __shfl_xor(sum, off, 32);
    const float inv = 1.0f / sum;
    float* aOut = attnOut + (head * SEQ + qbase + r) * SEQ;
    for (int k0 = lane; k0 < SEQ; k0 += 32) {
      const float p = Pr[k0] * inv;
      Pr[k0] = p;
      aOut[k0] = p;                      // the single mandatory attn writeback
    }
  }
  __syncthreads();

  // ---- phase 3: ctx = P @ V, split-K across wave pairs
  const int ntile = wave & 3;            // which 16 cols of d=64
  const int khalf = wave >> 2;           // which 1024 keys
  const bf16_t* Vcol = VhT + (head * DEPTH + ntile * 16 + l16) * SEQ;
  const float* Prow = &Pb[l16][0];
  v8f c = {};
  for (int kb = khalf * 1024; kb < khalf * 1024 + 1024; kb += 32) {
    const int k0 = kb + halfw * 16;
    const float4 f0 = *(const float4*)(Prow + k0);        // ds_load_b128
    const float4 f1 = *(const float4*)(Prow + k0 + 4);
    const float4 f2 = *(const float4*)(Prow + k0 + 8);
    const float4 f3 = *(const float4*)(Prow + k0 + 12);
    v16bf a;
    a[0]=(bf16_t)f0.x;  a[1]=(bf16_t)f0.y;  a[2]=(bf16_t)f0.z;  a[3]=(bf16_t)f0.w;
    a[4]=(bf16_t)f1.x;  a[5]=(bf16_t)f1.y;  a[6]=(bf16_t)f1.z;  a[7]=(bf16_t)f1.w;
    a[8]=(bf16_t)f2.x;  a[9]=(bf16_t)f2.y;  a[10]=(bf16_t)f2.z; a[11]=(bf16_t)f2.w;
    a[12]=(bf16_t)f3.x; a[13]=(bf16_t)f3.y; a[14]=(bf16_t)f3.z; a[15]=(bf16_t)f3.w;
    v16bf vb = load_bfrag(Vcol + kb + halfw * 16);
    c = wmma_bf16(a, vb, c);
  }
  if (wave >= 4) {
#pragma unroll
    for (int i = 0; i < 8; ++i) partC[ntile][halfw * 8 + i][l16] = c[i];
  }
  __syncthreads();
  if (wave < 4) {
#pragma unroll
    for (int i = 0; i < 8; ++i) {
      const int r = halfw * 8 + i;
      const float v = c[i] + partC[ntile][r][l16];
      const size_t m = (size_t)b * SEQ + qbase + r;
      Ctx[m * D_MODEL + h * DEPTH + ntile * 16 + l16] = (bf16_t)v;
    }
  }
}

// ---- host-side orchestration ------------------------------------------------

extern "C" void kernel_launch(void* const* d_in, const int* in_sizes, int n_in,
                              void* d_out, int out_size, void* d_ws, size_t ws_size,
                              hipStream_t stream)
{
  (void)in_sizes; (void)n_in; (void)out_size; (void)ws_size;
  const float*   q    = (const float*)d_in[0];
  const float*   k    = (const float*)d_in[1];
  const float*   v    = (const float*)d_in[2];
  const float*   Wq   = (const float*)d_in[3];
  const float*   bq   = (const float*)d_in[4];
  const float*   Wk   = (const float*)d_in[5];
  const float*   bk   = (const float*)d_in[6];
  const float*   Wv   = (const float*)d_in[7];
  const float*   bv   = (const float*)d_in[8];
  const float*   Wo   = (const float*)d_in[9];
  const float*   bo   = (const float*)d_in[10];
  const uint8_t* mask = (const uint8_t*)d_in[11];

  char* ws = (char*)d_ws;
  const size_t XSZ = (size_t)MROWS  * D_MODEL * sizeof(bf16_t);  // 8 MB
  const size_t WSZ = (size_t)D_MODEL * D_MODEL * sizeof(bf16_t); // 2 MB
  bf16_t* Xq  = (bf16_t*)(ws);
  bf16_t* Xk  = (bf16_t*)(ws + XSZ);
  bf16_t* Xv  = (bf16_t*)(ws + 2 * XSZ);
  bf16_t* WqT = (bf16_t*)(ws + 3 * XSZ);
  bf16_t* WkT = (bf16_t*)(ws + 3 * XSZ + WSZ);
  bf16_t* WvT = (bf16_t*)(ws + 3 * XSZ + 2 * WSZ);
  bf16_t* WoT = (bf16_t*)(ws + 3 * XSZ + 3 * WSZ);
  bf16_t* Qh  = (bf16_t*)(ws + 3 * XSZ + 4 * WSZ);
  bf16_t* Kh  = (bf16_t*)(ws + 4 * XSZ + 4 * WSZ);
  bf16_t* VhT = (bf16_t*)(ws + 5 * XSZ + 4 * WSZ);
  bf16_t* Ctx = (bf16_t*)(ws + 6 * XSZ + 4 * WSZ);   // total ws: 64 MB

  float* outP    = (float*)d_out;
  float* attnOut = outP + (size_t)MROWS * D_MODEL;   // tuple order: (out, attn)

  const int nX = MROWS * D_MODEL;
  cvt_f32_to_bf16<<<(nX + 255) / 256, 256, 0, stream>>>(q, Xq, nX);
  cvt_f32_to_bf16<<<(nX + 255) / 256, 256, 0, stream>>>(k, Xk, nX);
  cvt_f32_to_bf16<<<(nX + 255) / 256, 256, 0, stream>>>(v, Xv, nX);

  const int nW = D_MODEL * D_MODEL;
  transpose_cvt_1k<<<nW / 256, 256, 0, stream>>>(Wq, WqT);
  transpose_cvt_1k<<<nW / 256, 256, 0, stream>>>(Wk, WkT);
  transpose_cvt_1k<<<nW / 256, 256, 0, stream>>>(Wv, WvT);
  transpose_cvt_1k<<<nW / 256, 256, 0, stream>>>(Wo, WoT);

  dim3 g1(D_MODEL / 64, MROWS / 32);
  gemm_qkvo<<<g1, 256, 0, stream>>>(Xq, WqT, bq, (void*)Qh, 0);
  gemm_qkvo<<<g1, 256, 0, stream>>>(Xk, WkT, bk, (void*)Kh, 0);
  gemm_qkvo<<<g1, 256, 0, stream>>>(Xv, WvT, bv, (void*)VhT, 1);

  dim3 g2(SEQ / 16, NHEADS, BATCH);
  attn_fused<<<g2, 256, 0, stream>>>(Qh, Kh, VhT, mask, attnOut, Ctx);

  gemm_qkvo<<<g1, 256, 0, stream>>>(Ctx, WoT, bo, d_out, 2);
}
